// MPNN_6485400617637
// MI455X (gfx1250) — matrix operand order repaired and checked by reading, not compiled, not run on hardware
//
#include <hip/hip_runtime.h>

typedef __attribute__((ext_vector_type(8)))  float  v8f;
typedef __attribute__((ext_vector_type(16))) __bf16 v16bf;

struct alignas(32) U16x16 { unsigned short u[16]; };
struct alignas(32) U4x2   { uint4 lo; uint4 hi; };

#define DIM  128
#define TWOD 256

__device__ __forceinline__ unsigned short f2bf(float f) {
    unsigned int u = __builtin_bit_cast(unsigned int, f);
    u += 0x7FFFu + ((u >> 16) & 1u);           // round-to-nearest-even
    return (unsigned short)(u >> 16);
}

// pack two fp32 -> one dword of two bf16 (round-half-up) with a single v_perm_b32
__device__ __forceinline__ unsigned int pk_bf16(float flo, float fhi) {
    unsigned int a = __builtin_bit_cast(unsigned int, flo) + 0x8000u;
    unsigned int b = __builtin_bit_cast(unsigned int, fhi) + 0x8000u;
    // out = { b[31:16], a[31:16] }
    return __builtin_amdgcn_perm(b, a, 0x07060302u);
}

// ---------------- utility kernels ----------------
__global__ void k_zero(float* __restrict__ p, int n) {
    int i = blockIdx.x * 256 + threadIdx.x;
    if (i < n) p[i] = 0.f;
}

__global__ void k_deg(const int* __restrict__ colv, float* __restrict__ deg, int E_) {
    int i = blockIdx.x * 256 + threadIdx.x;
    if (i < E_) atomicAdd(&deg[colv[i]], 1.0f);
}

__global__ void k_dinv(float* __restrict__ deg, int N_) {
    int i = blockIdx.x * 256 + threadIdx.x;
    if (i < N_) { float d = deg[i]; deg[i] = (d > 0.f) ? rsqrtf(d) : 0.f; }
}

__global__ void k_norm(const int* __restrict__ rowv, const int* __restrict__ colv,
                       const float* __restrict__ dinv, float* __restrict__ norm, int E_) {
    int i = blockIdx.x * 256 + threadIdx.x;
    if (i < E_) norm[i] = dinv[rowv[i]] * dinv[colv[i]];
}

// Pack Wm (fp32 [L, D, 2D]) into bf16 B-fragments (v_wmma_f32_16x16x32_bf16
// B layout): entry = ((l*8 + w)*8 + kb)*32 + lane; element i holds
// B[k = kb*32 + (i < 8 ? half*8 + i : 16 + half*8 + i - 8)][n = w*16 + (lane&15)] with B[k][n] = Wm[n][k].
__global__ void k_packw(const float* __restrict__ Wm, unsigned short* __restrict__ wpack, int Ltot) {
    int idx = blockIdx.x * 256 + threadIdx.x;
    int total = Ltot * 8 * 8 * 32;
    if (idx >= total) return;
    int lane = idx & 31;
    int kb   = (idx >> 5) & 7;
    int w    = (idx >> 8) & 7;
    int l    = idx >> 11;
    int half = lane >> 4, lcol = lane & 15;
    int n = w * 16 + lcol;
    const float* src = Wm + ((long)l * DIM + n) * TWOD;
    unsigned short* dst = wpack + (long)idx * 16;
#pragma unroll
    for (int i = 0; i < 16; ++i) {
        int k = kb * 32 + (i < 8 ? half * 8 + i : 16 + half * 8 + (i - 8));
        dst[i] = f2bf(src[k]);
    }
}

// Edge message + scatter: one wave (32 lanes) per edge, 4 features per lane.
__global__ void __launch_bounds__(256)
k_edge(const float* __restrict__ h_in, const int* __restrict__ rowv,
       const int* __restrict__ colv, const float* __restrict__ attr,
       const float* __restrict__ norm, const float* __restrict__ W1l,
       const float* __restrict__ W2l, float* __restrict__ agg, int E_) {
    int idx = blockIdx.x * 256 + threadIdx.x;
    int e = idx >> 5;
    if (e >= E_) return;
    int d = (idx & 31) * 4;
    int r = rowv[e], c = colv[e];
    float nv = norm[e];
    float4 a  = *(const float4*)(attr + (long)e * 4);
    float4 w0 = *(const float4*)(W1l + d);
    float4 w1 = *(const float4*)(W1l + DIM + d);
    float4 w2 = *(const float4*)(W1l + 2 * DIM + d);
    float4 w3 = *(const float4*)(W2l + d);
    float4 h  = *(const float4*)(h_in + (long)r * DIM + d);
    float m0 = nv * (h.x + a.x * w0.x + a.y * w1.x + a.z * w2.x + a.w * w3.x);
    float m1 = nv * (h.y + a.x * w0.y + a.y * w1.y + a.z * w2.y + a.w * w3.y);
    float m2 = nv * (h.z + a.x * w0.z + a.y * w1.z + a.z * w2.z + a.w * w3.z);
    float m3 = nv * (h.w + a.x * w0.w + a.y * w1.w + a.z * w2.w + a.w * w3.w);
    float* dst = agg + (long)c * DIM + d;
    atomicAdd(dst + 0, m0);
    atomicAdd(dst + 1, m1);
    atomicAdd(dst + 2, m2);
    atomicAdd(dst + 3, m3);
}

// WMMA GEMM: h2 = relu([h | agg] @ Wm^T + b) with fused BN sum/sum^2.
// 256 threads = 8 waves. Block owns 128 rows x all 128 columns.
// Phase 1: block cooperatively converts the 128x256 fp32 A panel to bf16 and
//          stores it in LDS pre-swizzled into WMMA A-fragment layout (64 KB);
//          each global element read exactly once per block.
// Phase 2: wave w computes columns [16w,16w+16); A fragments via 2x
//          ds_load_b128, B fragments register-resident from packed global.
__global__ void __launch_bounds__(256)
k_gemm(const float* __restrict__ h_in, const float* __restrict__ agg,
       const unsigned short* __restrict__ wpackl, const float* __restrict__ biasl,
       float* __restrict__ h_out, float* __restrict__ bnsum,
       float* __restrict__ bnsq, int nrows) {
    // fragment (t,kb): 1024 B block; first 512 B = elems 0..7 (lane*16),
    // second 512 B = elems 8..15 (lane*16)
    __shared__ unsigned char smem[8 * 8 * 1024];   // 64 KB

    const int tid  = threadIdx.x;
    const int wave = tid >> 5;
    const int lane = tid & 31;
    const int half = lane >> 4;
    const int lcol = lane & 15;
    const int n0   = wave * 16;
    const int m_base = blockIdx.x * 128;

    // preload all 8 B fragments (reused by 8 row-subtiles)
    const U16x16* wp = (const U16x16*)wpackl + (wave * 8) * 32 + lane;
    v16bf bfrag[8];
#pragma unroll
    for (int kb = 0; kb < 8; ++kb)
        bfrag[kb] = __builtin_bit_cast(v16bf, wp[kb * 32]);

    const float bias = biasl[n0 + lcol];

    // ---- Phase 1: stage A panel into LDS in fragment layout ----
    {
        const int kb_s = tid >> 5;       // 0..7, fixed per thread
        const int m_s  = lane & 15;      // row within subtile
#pragma unroll
        for (int t = 0; t < 8; ++t) {
            int row = m_base + t * 16 + m_s;
            if (row >= nrows) row = nrows - 1;   // clamped; garbage rows masked at store
            const float* src = (kb_s < 4) ? (h_in + (long)row * DIM + kb_s * 32)
                                          : (agg  + (long)row * DIM + (kb_s - 4) * 32);
            // A layout: half 0 -> K {0..7, 16..23}; half 1 -> K {8..15, 24..31}
            float4 f0 = *(const float4*)(src + half * 8);
            float4 f1 = *(const float4*)(src + half * 8 + 4);
            float4 f2 = *(const float4*)(src + 16 + half * 8);
            float4 f3 = *(const float4*)(src + 16 + half * 8 + 4);
            uint4 lo, hi;
            lo.x = pk_bf16(f0.x, f0.y);  lo.y = pk_bf16(f0.z, f0.w);
            lo.z = pk_bf16(f1.x, f1.y);  lo.w = pk_bf16(f1.z, f1.w);
            hi.x = pk_bf16(f2.x, f2.y);  hi.y = pk_bf16(f2.z, f2.w);
            hi.z = pk_bf16(f3.x, f3.y);  hi.w = pk_bf16(f3.z, f3.w);
            unsigned char* fb = smem + (t * 8 + kb_s) * 1024;
            *(uint4*)(fb + lane * 16)       = lo;
            *(uint4*)(fb + 512 + lane * 16) = hi;
        }
    }
    __syncthreads();

    // ---- Phase 2: WMMA over 8 row-subtiles ----
    for (int t = 0; t < 8; ++t) {
        int m0 = m_base + t * 16;

        v8f c = {};
#pragma unroll
        for (int kb = 0; kb < 8; ++kb) {
            const unsigned char* fb = smem + (t * 8 + kb) * 1024;
            U4x2 au;
            au.lo = *(const uint4*)(fb + lane * 16);
            au.hi = *(const uint4*)(fb + 512 + lane * 16);
            v16bf afrag = __builtin_bit_cast(v16bf, au);
            c = __builtin_amdgcn_wmma_f32_16x16x32_bf16(
                    false, afrag, false, bfrag[kb], (short)0, c, false, false);
        }

        // epilogue: bias + ReLU, store, fused BN partial sums.
        // Wave-uniform bounds branch: only the tail subtiles of the final
        // block take the masked path, so the hot path is straight-line with
        // one base address and immediate-offset stores.
        float s = 0.f, q = 0.f;
        if (m0 + 16 <= nrows) {
            float* p = h_out + (long)(m0 + half * 8) * DIM + n0 + lcol;
#pragma unroll
            for (int r = 0; r < 8; ++r) {
                float v = fmaxf(c[r] + bias, 0.f);
                p[(long)r * DIM] = v;            // row stride 512 B -> inst offsets
                s += v;
                q = fmaf(v, v, q);
            }
        } else {
#pragma unroll
            for (int r = 0; r < 8; ++r) {
                int rowm = m0 + half * 8 + r;    // C layout: VGPR r -> row r + 8*half
                float v = fmaxf(c[r] + bias, 0.f);
                if (rowm < nrows) {
                    h_out[(long)rowm * DIM + n0 + lcol] = v;
                    s += v;
                    q = fmaf(v, v, q);
                }
            }
        }
        atomicAdd(&bnsum[n0 + lcol], s);
        atomicAdd(&bnsq[n0 + lcol], q);
    }
}

__global__ void k_bnfin(const float* __restrict__ bnsum, const float* __restrict__ bnsq,
                        const float* __restrict__ gamma, const float* __restrict__ beta,
                        float* __restrict__ scale, float* __restrict__ shift, float invN) {
    int n = threadIdx.x;  // 128 threads
    float mean = bnsum[n] * invN;
    float var  = bnsq[n] * invN - mean * mean;
    float sc   = gamma[n] * rsqrtf(var + 1e-5f);
    scale[n] = sc;
    shift[n] = beta[n] - mean * sc;
}

__global__ void k_bnapply(float* __restrict__ h, const float* __restrict__ scale,
                          const float* __restrict__ shift, int total4, int relu) {
    int i = blockIdx.x * 256 + threadIdx.x;
    if (i >= total4) return;
    int d = (i & 31) * 4;
    float4 v  = *(float4*)(h + (long)i * 4);
    float4 sc = *(const float4*)(scale + d);
    float4 sh = *(const float4*)(shift + d);
    v.x = v.x * sc.x + sh.x;
    v.y = v.y * sc.y + sh.y;
    v.z = v.z * sc.z + sh.z;
    v.w = v.w * sc.w + sh.w;
    if (relu) {
        v.x = fmaxf(v.x, 0.f); v.y = fmaxf(v.y, 0.f);
        v.z = fmaxf(v.z, 0.f); v.w = fmaxf(v.w, 0.f);
    }
    *(float4*)(h + (long)i * 4) = v;
}

extern "C" void kernel_launch(void* const* d_in, const int* in_sizes, int n_in,
                              void* d_out, int out_size, void* d_ws, size_t ws_size,
                              hipStream_t stream) {
    const float* x     = (const float*)d_in[0];
    const int*   eidx  = (const int*)d_in[1];
    const float* eattr = (const float*)d_in[2];
    const float* W1    = (const float*)d_in[3];
    const float* W2    = (const float*)d_in[4];
    const float* Wm    = (const float*)d_in[5];
    const float* bm    = (const float*)d_in[6];
    const float* gamma = (const float*)d_in[7];
    const float* beta  = (const float*)d_in[8];

    const int N = in_sizes[0] / DIM;
    const int E = in_sizes[1] / 2;
    const int L = in_sizes[3] / (3 * DIM);
    const int* rowv = eidx;
    const int* colv = eidx + E;

    // workspace carve-out (256B aligned)
    char* ws = (char*)d_ws;
    size_t o = 0;
    auto carve = [&](size_t bytes) -> void* {
        void* p = ws + o;
        o = (o + bytes + 255) & ~(size_t)255;
        return p;
    };
    float* deg     = (float*)carve((size_t)N * 4);
    float* norm    = (float*)carve((size_t)E * 4);
    float* bufA    = (float*)carve((size_t)N * DIM * 4);
    float* bufB    = (float*)carve((size_t)N * DIM * 4);
    float* agg     = (float*)carve((size_t)N * DIM * 4);
    float* bnsum   = (float*)carve(2 * DIM * 4);   // sum | sumsq contiguous
    float* bnsq    = bnsum + DIM;
    float* bnscale = (float*)carve(2 * DIM * 4);
    float* bnshift = bnscale + DIM;
    unsigned short* wpack = (unsigned short*)carve((size_t)L * 8 * 8 * 32 * 16 * 2);

    dim3 blk(256);
    k_zero<<<(N + 255) / 256, blk, 0, stream>>>(deg, N);
    k_deg <<<(E + 255) / 256, blk, 0, stream>>>(colv, deg, E);
    k_dinv<<<(N + 255) / 256, blk, 0, stream>>>(deg, N);
    k_norm<<<(E + 255) / 256, blk, 0, stream>>>(rowv, colv, deg, norm, E);

    int packTot = L * 8 * 8 * 32;
    k_packw<<<(packTot + 255) / 256, blk, 0, stream>>>(Wm, wpack, L);

    float* out = (float*)d_out;
    const float* h_in = x;
    for (int l = 0; l < L; ++l) {
        float* h_out = (l == L - 1) ? out : ((l % 2 == 0) ? bufA : bufB);

        k_zero<<<((N * DIM) + 255) / 256, blk, 0, stream>>>(agg, N * DIM);
        k_zero<<<1, blk, 0, stream>>>(bnsum, 2 * DIM);

        k_edge<<<(E * 32 + 255) / 256, blk, 0, stream>>>(
            h_in, rowv, colv, eattr, norm,
            W1 + (size_t)l * 3 * DIM, W2 + (size_t)l * DIM, agg, E);

        k_gemm<<<(N + 127) / 128, blk, 0, stream>>>(
            h_in, agg, wpack + (size_t)l * 8 * 8 * 32 * 16,
            bm + (size_t)l * DIM, h_out, bnsum, bnsq, N);

        k_bnfin<<<1, DIM, 0, stream>>>(bnsum, bnsq, gamma + (size_t)l * DIM,
                                       beta + (size_t)l * DIM, bnscale, bnshift,
                                       1.0f / (float)N);

        k_bnapply<<<((N * DIM / 4) + 255) / 256, blk, 0, stream>>>(
            h_out, bnscale, bnshift, N * DIM / 4, (l < L - 1) ? 1 : 0);

        h_in = h_out;
    }
}
